// Graph_Layer_44787918963015
// MI455X (gfx1250) — matrix operand order repaired and checked by reading, not compiled, not run on hardware
//
#include <hip/hip_runtime.h>
#include <hip/hip_bf16.h>

// ---------- types ----------
typedef __attribute__((ext_vector_type(16))) __bf16 v16bf;
typedef __attribute__((ext_vector_type(8)))  float  v8f;

static __device__ __forceinline__ v16bf make_frag(uint4 lo, uint4 hi) {
    union { uint4 u[2]; v16bf v; } t;
    t.u[0] = lo; t.u[1] = hi;
    return t.v;
}

static __device__ __forceinline__ void cvt4_store(__bf16* dst, float4 f) {
    union { __bf16 h[4]; uint2 u; } t;
    t.h[0] = (__bf16)f.x; t.h[1] = (__bf16)f.y;
    t.h[2] = (__bf16)f.z; t.h[3] = (__bf16)f.w;
    *(uint2*)dst = t.u;
}

// async 16-byte copy global -> LDS, tracked by ASYNCcnt (CDNA5)
static __device__ __forceinline__ void async_copy_b128(unsigned lds_off, const void* gaddr) {
    asm volatile("global_load_async_to_lds_b128 %0, %1, off"
                 :: "v"(lds_off), "v"(gaddr) : "memory");
}

#define WMMA_BF16(A, B, C) \
    __builtin_amdgcn_wmma_f32_16x16x32_bf16(false, (A), false, (B), (short)0, (C), false, false)

// =====================================================================
// Kernel 0: sim_feat fp32 -> bf16 (one-time, makes hot-loop staging a
// pure async byte copy and halves K-stream L2 traffic)
// =====================================================================
__global__ void cvt_sim_kernel(const float* __restrict__ sim,
                               __bf16* __restrict__ simb) {
    int i = blockIdx.x * blockDim.x + threadIdx.x;   // one float4 each
    float4 f = *(const float4*)(sim + (size_t)i * 4);
    cvt4_store(simb + (size_t)i * 4, f);
}

// =====================================================================
// Kernel 1: V' = x @ W, stored TRANSPOSED as bf16: vt[j*8192 + n]
// grid (64, 8), block 256 (8 waves); each wave: 16 rows x 64 cols
// =====================================================================
__global__ void xw_gemm_kernel(const float* __restrict__ x,
                               const float* __restrict__ w,
                               __bf16* __restrict__ vt) {
    extern __shared__ char smem[];
    __bf16* wt = (__bf16*)smem;                 // [64 j][512 d] bf16 = 64 KB
    const int nb = blockIdx.x;
    const int jb = blockIdx.y;
    const int tid = threadIdx.x;

    for (int i = tid; i < 64 * 512; i += 256) {
        int d = i >> 6;
        int j = i & 63;
        wt[j * 512 + d] = (__bf16)w[d * 512 + jb * 64 + j];
    }
    __syncthreads();

    const int lane = tid & 31;
    const int wv   = tid >> 5;
    const int ln   = lane & 15;
    const int lg   = lane >> 4;
    const int row0 = nb * 128 + wv * 16;

    v8f acc[4];
#pragma unroll
    for (int ct = 0; ct < 4; ++ct)
#pragma unroll
        for (int r = 0; r < 8; ++r) acc[ct][r] = 0.0f;

#pragma unroll
    for (int d0 = 0; d0 < 512; d0 += 32) {
        const float* xp = x + (row0 + ln) * 512 + d0 + 8 * lg;
        float4 f0 = *(const float4*)(xp);
        float4 f1 = *(const float4*)(xp + 4);
        float4 f2 = *(const float4*)(xp + 16);
        float4 f3 = *(const float4*)(xp + 20);
        v16bf a;
        a[0]=(__bf16)f0.x; a[1]=(__bf16)f0.y; a[2]=(__bf16)f0.z; a[3]=(__bf16)f0.w;
        a[4]=(__bf16)f1.x; a[5]=(__bf16)f1.y; a[6]=(__bf16)f1.z; a[7]=(__bf16)f1.w;
        a[8]=(__bf16)f2.x; a[9]=(__bf16)f2.y; a[10]=(__bf16)f2.z; a[11]=(__bf16)f2.w;
        a[12]=(__bf16)f3.x; a[13]=(__bf16)f3.y; a[14]=(__bf16)f3.z; a[15]=(__bf16)f3.w;

#pragma unroll
        for (int ct = 0; ct < 4; ++ct) {
            const __bf16* bp = wt + (ct * 16 + ln) * 512 + d0 + 8 * lg;
            v16bf b = make_frag(*(const uint4*)bp, *(const uint4*)(bp + 16));
            acc[ct] = WMMA_BF16(a, b, acc[ct]);
        }
    }

#pragma unroll
    for (int ct = 0; ct < 4; ++ct) {
        int j = jb * 64 + ct * 16 + ln;
#pragma unroll
        for (int r = 0; r < 8; ++r) {
            int n = row0 + r + 8 * lg;
            vt[j * 8192 + n] = (__bf16)acc[ct][r];
        }
    }
}

// =====================================================================
// Kernel 2: flash-attention  out = softmax(sim.simT) @ V'
// grid 128, block 128 (4 waves). Each wave: 16 query rows x 512 cols,
// online softmax, 16x512 fp32 accumulator in VGPRs.
// All tile staging via GLOBAL_LOAD_ASYNC_TO_LDS_B128 (ASYNCcnt).
// Dynamic LDS: Q 64KB | K 32KB | V 32KB | P 4KB = 132 KB -> 2 WG/WGP
// =====================================================================
__global__ void attn_kernel(const __bf16* __restrict__ simb,
                            const __bf16* __restrict__ vt,
                            float* __restrict__ out) {
    extern __shared__ char smem[];
    __bf16* qlds = (__bf16*)smem;                   // [64][512]
    __bf16* klds = (__bf16*)(smem + 65536);         // [32][512]
    __bf16* vlds = (__bf16*)(smem + 98304);         // [512 col][32 key]
    __bf16* pbuf = (__bf16*)(smem + 131072);        // [4 wave][16][32]

    const int qb   = blockIdx.x;
    const int tid  = threadIdx.x;
    const int lane = tid & 31;
    const int wv   = tid >> 5;
    const int ln   = lane & 15;
    const int lg   = lane >> 4;

    const unsigned q_off = (unsigned)(uintptr_t)qlds;
    const unsigned k_off = (unsigned)(uintptr_t)klds;
    const unsigned v_off = (unsigned)(uintptr_t)vlds;

    // stage Q block asynchronously: 64 rows x 1KB = 4096 x 16B
    {
        const char* qsrc = (const char*)(simb + (size_t)qb * 64 * 512);
        for (int i = tid; i < 4096; i += 128)
            async_copy_b128(q_off + i * 16, qsrc + (size_t)i * 16);
    }

    float m[8], l[8];
    v8f acc[32];
#pragma unroll
    for (int r = 0; r < 8; ++r) { m[r] = -__builtin_inff(); l[r] = 0.0f; }
#pragma unroll
    for (int ct = 0; ct < 32; ++ct)
#pragma unroll
        for (int r = 0; r < 8; ++r) acc[ct][r] = 0.0f;

    __bf16* mypb = pbuf + wv * 512;

    for (int kb = 0; kb < 8192; kb += 32) {
        __syncthreads();                 // all waves done reading prev K/V

        // stage K tile: 32 keys x 1KB = 2048 x 16B (pure async byte copy)
        {
            const char* ksrc = (const char*)(simb + (size_t)kb * 512);
            for (int i = tid; i < 2048; i += 128)
                async_copy_b128(k_off + i * 16, ksrc + (size_t)i * 16);
        }
        // stage V' tile: 512 cols x 64B (already transposed bf16 in ws)
        for (int i = tid; i < 2048; i += 128) {
            int c  = i >> 2;
            int q4 = i & 3;
            async_copy_b128(v_off + (unsigned)(c * 64 + q4 * 16),
                            vt + (size_t)c * 8192 + kb + q4 * 8);
        }
        asm volatile("s_wait_asynccnt 0" ::: "memory");
        __syncthreads();                 // tiles visible to all waves

        // ---- S = Q . K^T : 4 independent WMMA chains (2 key tiles x 2 d-halves)
        v8f s0a, s0b, s1a, s1b;
#pragma unroll
        for (int r = 0; r < 8; ++r) { s0a[r]=0.f; s0b[r]=0.f; s1a[r]=0.f; s1b[r]=0.f; }

#pragma unroll
        for (int d0 = 0; d0 < 256; d0 += 32) {
            const __bf16* qp = qlds + (wv * 16 + ln) * 512 + d0 + 16 * lg;
            v16bf a_lo = make_frag(*(const uint4*)qp,         *(const uint4*)(qp + 8));
            v16bf a_hi = make_frag(*(const uint4*)(qp + 256), *(const uint4*)(qp + 264));

            const __bf16* k0 = klds + ln * 512 + d0 + 16 * lg;
            const __bf16* k1 = klds + (16 + ln) * 512 + d0 + 16 * lg;
            v16bf b0_lo = make_frag(*(const uint4*)k0,         *(const uint4*)(k0 + 8));
            v16bf b0_hi = make_frag(*(const uint4*)(k0 + 256), *(const uint4*)(k0 + 264));
            v16bf b1_lo = make_frag(*(const uint4*)k1,         *(const uint4*)(k1 + 8));
            v16bf b1_hi = make_frag(*(const uint4*)(k1 + 256), *(const uint4*)(k1 + 264));

            s0a = WMMA_BF16(a_lo, b0_lo, s0a);
            s1a = WMMA_BF16(a_lo, b1_lo, s1a);
            s0b = WMMA_BF16(a_hi, b0_hi, s0b);
            s1b = WMMA_BF16(a_hi, b1_hi, s1b);
        }
        v8f st0, st1;
#pragma unroll
        for (int r = 0; r < 8; ++r) { st0[r] = s0a[r] + s0b[r]; st1[r] = s1a[r] + s1b[r]; }

        // ---- online softmax (row r+8*lg lives in a 16-lane group) ----
        float tm[8], p0[8], p1[8], rs[8], sc[8];
#pragma unroll
        for (int r = 0; r < 8; ++r) tm[r] = fmaxf(st0[r], st1[r]);
#pragma unroll
        for (int r = 0; r < 8; ++r) {
#pragma unroll
            for (int mk = 1; mk < 16; mk <<= 1)
                tm[r] = fmaxf(tm[r], __shfl_xor(tm[r], mk, 32));
        }
#pragma unroll
        for (int r = 0; r < 8; ++r) {
            float mn = fmaxf(m[r], tm[r]);
            sc[r] = __expf(m[r] - mn);
            p0[r] = __expf(st0[r] - mn);
            p1[r] = __expf(st1[r] - mn);
            m[r]  = mn;
            rs[r] = p0[r] + p1[r];
        }
#pragma unroll
        for (int r = 0; r < 8; ++r) {
#pragma unroll
            for (int mk = 1; mk < 16; mk <<= 1)
                rs[r] += __shfl_xor(rs[r], mk, 32);
            l[r] = l[r] * sc[r] + rs[r];
        }
#pragma unroll
        for (int ct = 0; ct < 32; ++ct)
#pragma unroll
            for (int r = 0; r < 8; ++r) acc[ct][r] *= sc[r];

        // ---- C-layout P -> A-fragment via wave-private LDS ----
#pragma unroll
        for (int r = 0; r < 8; ++r) {
            mypb[(r + 8 * lg) * 32 + ln]      = (__bf16)p0[r];
            mypb[(r + 8 * lg) * 32 + 16 + ln] = (__bf16)p1[r];
        }
        asm volatile("s_wait_dscnt 0" ::: "memory");   // cross-lane LDS RAW (same wave)

        const __bf16* pp = mypb + ln * 32 + 16 * lg;
        v16bf pf = make_frag(*(const uint4*)pp, *(const uint4*)(pp + 8));

        // ---- acc += P @ V' over all 512 output columns (32 chains) ----
#pragma unroll
        for (int ct = 0; ct < 32; ++ct) {
            const __bf16* vp = vlds + (ct * 16 + ln) * 32 + 16 * lg;
            v16bf b = make_frag(*(const uint4*)vp, *(const uint4*)(vp + 8));
            acc[ct] = WMMA_BF16(pf, b, acc[ct]);
        }
    }

    // ---- epilogue: normalize and store fp32 ----
    float rl[8];
#pragma unroll
    for (int r = 0; r < 8; ++r) rl[r] = 1.0f / l[r];
#pragma unroll
    for (int ct = 0; ct < 32; ++ct) {
        int col = ct * 16 + ln;
#pragma unroll
        for (int r = 0; r < 8; ++r) {
            int row = qb * 64 + wv * 16 + r + 8 * lg;
            out[row * 512 + col] = acc[ct][r] * rl[r];
        }
    }
}

// =====================================================================
extern "C" void kernel_launch(void* const* d_in, const int* in_sizes, int n_in,
                              void* d_out, int out_size, void* d_ws, size_t ws_size,
                              hipStream_t stream) {
    const float* x   = (const float*)d_in[0];   // [8192, 512]
    const float* sim = (const float*)d_in[1];   // [8192, 512]
    const float* w   = (const float*)d_in[2];   // [512, 512]
    float* out = (float*)d_out;                 // [8192, 512]

    __bf16* vt   = (__bf16*)d_ws;                          // V'^T [512][8192] bf16, 8 MB
    __bf16* simb = (__bf16*)((char*)d_ws + (8u << 20));    // sim bf16 [8192][512], 8 MB

    cvt_sim_kernel<<<4096, 256, 0, stream>>>(sim, simb);
    xw_gemm_kernel<<<dim3(64, 8), dim3(256), 64 * 1024, stream>>>(x, w, vt);
    attn_kernel<<<dim3(128), dim3(128), 132 * 1024, stream>>>(simb, vt, out);
}
